// FeatureQuantizerEMA_30932354466466
// MI455X (gfx1250) — compile-verified, matrix operand order checked by reading, not run on hardware
//
#include <hip/hip_runtime.h>
#include <hip/hip_fp16.h>

// ---------------------------------------------------------------------------
// FeatureQuantizerEMA for MI455X (gfx1250, wave32, WMMA)
// Outputs (flat f32, concatenated in reference return order):
//   loss[1], quant_st[16*1024*3*256], perplexity[1], avg_probs[4096],
//   encoding_indices[16*1024*3], distances[16384*4096], new_cs[4096],
//   new_ema_w[4096*256], updated_embeddings[4096*256]
// ---------------------------------------------------------------------------

typedef __attribute__((ext_vector_type(16))) _Float16 v16h;
typedef __attribute__((ext_vector_type(8)))  _Float16 v8h;
typedef __attribute__((ext_vector_type(8)))  float    v8f;
typedef int v4i __attribute__((vector_size(16)));

static constexpr int   Bv   = 16;
static constexpr int   Tv   = 1024;
static constexpr int   Dv   = 256;
static constexpr int   Kv   = 4096;
static constexpr int   Nv   = Bv * Tv;     // 16384
static constexpr int   TOPK = 3;
static constexpr int   NCTILE = Kv / 128;  // 32 column tiles
static constexpr int   CSLOTS = NCTILE * 6;  // 192 candidates per row
static constexpr float DECAYF = 0.99f;
static constexpr float EPSF   = 1e-5f;
static constexpr float CCOST  = 0.25f;

// CDNA5 async global->LDS copies (ASYNCcnt), with safe fallback.
#if defined(__has_builtin)
#if __has_builtin(__builtin_amdgcn_global_load_async_to_lds_b128) && \
    __has_builtin(__builtin_amdgcn_s_wait_asynccnt)
#define USE_ASYNC_COPY 1
#endif
#endif
// Builtin signature (from hipcc diagnostics): (int4 AS1*, int4 AS3*, imm, imm)
#define GPTR(p) ((__attribute__((address_space(1))) v4i*)(p))
#define LPTR(p) ((__attribute__((address_space(3))) v4i*)(p))

// ---------------------------------------------------------------------------
// Kernel 1: fp32 -> fp16 conversion + squared row norms. One wave per row.
// ---------------------------------------------------------------------------
__global__ __launch_bounds__(256) void k_convert(
    const float* __restrict__ x, const float* __restrict__ e,
    _Float16* __restrict__ xh, _Float16* __restrict__ eh,
    float* __restrict__ xnorm, float* __restrict__ enorm) {
  const int wave = (int)((blockIdx.x * blockDim.x + threadIdx.x) >> 5);
  const int lane = threadIdx.x & 31;
  const float* src;
  _Float16* dst;
  float* nrm;
  if (wave < Nv) {
    src = x + (size_t)wave * Dv;  dst = xh + (size_t)wave * Dv;  nrm = xnorm + wave;
  } else {
    const int row = wave - Nv;
    if (row >= Kv) return;
    src = e + (size_t)row * Dv;   dst = eh + (size_t)row * Dv;   nrm = enorm + row;
  }
  const float4* s4 = (const float4*)(const void*)src;
  const float4 a = s4[lane * 2 + 0];
  const float4 b = s4[lane * 2 + 1];
  v8h hv;
  hv[0] = (_Float16)a.x; hv[1] = (_Float16)a.y; hv[2] = (_Float16)a.z; hv[3] = (_Float16)a.w;
  hv[4] = (_Float16)b.x; hv[5] = (_Float16)b.y; hv[6] = (_Float16)b.z; hv[7] = (_Float16)b.w;
  ((v8h*)(void*)dst)[lane] = hv;
  float s = a.x * a.x + a.y * a.y + a.z * a.z + a.w * a.w
          + b.x * b.x + b.y * b.y + b.z * b.z + b.w * b.w;
  #pragma unroll
  for (int m = 16; m; m >>= 1) s += __shfl_xor(s, m, 32);
  if (lane == 0) *nrm = s;
}

// ---------------------------------------------------------------------------
// Kernel 2: zero scratch (count | dw | loss accumulator)
// ---------------------------------------------------------------------------
__global__ void k_zero(float* __restrict__ p, size_t n) {
  size_t i = (size_t)blockIdx.x * blockDim.x + threadIdx.x;
  if (i < n) p[i] = 0.0f;
}

// ---------------------------------------------------------------------------
// Top-3 insertion with duplicate-index guard (for butterfly merges).
// ---------------------------------------------------------------------------
__device__ __forceinline__ void top3_insert(float d, int i,
    float& d0, int& i0, float& d1, int& i1, float& d2, int& i2) {
  if (i == i0 || i == i1 || i == i2) return;
  if (d < d2) {
    if (d < d1) {
      d2 = d1; i2 = i1;
      if (d < d0) { d1 = d0; i1 = i0; d0 = d; i0 = i; }
      else        { d1 = d;  i1 = i; }
    } else { d2 = d; i2 = i; }
  }
}

// ---------------------------------------------------------------------------
// Kernel 3: distance GEMM, WMMA f32 <- f16 x f16, with fused per-tile top-3.
// Block: 256 threads (8 waves, 4x2 wave grid). Tile: 128 rows x 128 codes.
// Each wave: 32x64 subtile = 2x4 fragments of v_wmma_f32_16x16x32_f16.
// dists[n,k] = |x_n|^2 + |e_k|^2 - 2 * <x_n, e_k>
// Epilogue: stream dists out once + emit top-3 candidates per (row, 64-col
// slab) so the final top-k never re-reads the 268 MB distance matrix.
// ---------------------------------------------------------------------------
__global__ __launch_bounds__(256) void k_dists(
    const _Float16* __restrict__ xh, const _Float16* __restrict__ eh,
    const float* __restrict__ xnorm, const float* __restrict__ enorm,
    float* __restrict__ dists, float2* __restrict__ cand) {
  // 96-byte row stride -> every 16-half fragment is 32B aligned
  __shared__ _Float16 lA[128 * 48];
  __shared__ _Float16 lB[128 * 48];

  const int tid  = threadIdx.x;
  const int lane = tid & 31;
  const int wave = tid >> 5;
  const int wm   = wave >> 1;        // 0..3 : 32-row slab
  const int wn   = wave & 1;         // 0..1 : 64-col slab
  const int rowBase = blockIdx.y * 128;
  const int colBase = blockIdx.x * 128;

  v8f acc[2][4] = {};

  const int lr = tid >> 1;           // staging: row within tile (0..127)
  const int lp = (tid & 1) * 16;     // staging: 16-half slot within 32-half slab

  for (int kk = 0; kk < Dv; kk += 32) {
    __syncthreads();
    {
      const _Float16* ga = xh + (size_t)(rowBase + lr) * Dv + kk + lp;
      _Float16* sa = lA + lr * 48 + lp;
      const _Float16* gb = eh + (size_t)(colBase + lr) * Dv + kk + lp;
      _Float16* sb = lB + lr * 48 + lp;
#ifdef USE_ASYNC_COPY
      __builtin_amdgcn_global_load_async_to_lds_b128(GPTR(ga), LPTR(sa), 0, 0);
      __builtin_amdgcn_global_load_async_to_lds_b128(GPTR(ga), LPTR(sa), 16, 0);
      __builtin_amdgcn_global_load_async_to_lds_b128(GPTR(gb), LPTR(sb), 0, 0);
      __builtin_amdgcn_global_load_async_to_lds_b128(GPTR(gb), LPTR(sb), 16, 0);
#else
      const uint4* ga4 = (const uint4*)(const void*)ga;
      uint4* sa4 = (uint4*)(void*)sa;
      sa4[0] = ga4[0]; sa4[1] = ga4[1];
      const uint4* gb4 = (const uint4*)(const void*)gb;
      uint4* sb4 = (uint4*)(void*)sb;
      sb4[0] = gb4[0]; sb4[1] = gb4[1];
#endif
      if (kk + 32 < Dv) {  // pull next slab toward L2 (global_prefetch_b8)
        __builtin_prefetch(ga + 32, 0, 1);
        __builtin_prefetch(gb + 32, 0, 1);
      }
#ifdef USE_ASYNC_COPY
      __builtin_amdgcn_s_wait_asynccnt(0);
#endif
    }
    __syncthreads();

    const int khalf = (lane >> 4) << 4;  // 0 or 16: lanes 0-15 low K, 16-31 high K
    v16h afrag[2];
    #pragma unroll
    for (int mf = 0; mf < 2; mf++) {
      const int r = wm * 32 + mf * 16 + (lane & 15);
      afrag[mf] = *(const v16h*)(const void*)(lA + r * 48 + khalf);
    }
    #pragma unroll
    for (int nf = 0; nf < 4; nf++) {
      const int c = wn * 64 + nf * 16 + (lane & 15);
      v16h bfrag = *(const v16h*)(const void*)(lB + c * 48 + khalf);
      #pragma unroll
      for (int mf = 0; mf < 2; mf++) {
        acc[mf][nf] = __builtin_amdgcn_wmma_f32_16x16x32_f16(
            false, afrag[mf], false, bfrag, (short)0, acc[mf][nf], false, false);
      }
    }
  }

  // Epilogue A: stream the distance tile out.
  // C layout: lanes 0-15 -> M=r, lanes 16-31 -> M=8+r, N=lane%16.
  const int lcol = lane & 15;
  const int rsel = (lane < 16) ? 0 : 8;
  #pragma unroll
  for (int mf = 0; mf < 2; mf++) {
    const int r0 = rowBase + wm * 32 + mf * 16 + rsel;
    #pragma unroll
    for (int nf = 0; nf < 4; nf++) {
      const int c = colBase + wn * 64 + nf * 16 + lcol;
      const float en = enorm[c];
      #pragma unroll
      for (int r = 0; r < 8; r++) {
        dists[(size_t)(r0 + r) * Kv + c] = xnorm[r0 + r] + en - 2.0f * acc[mf][nf][r];
      }
    }
  }

  // Epilogue B: fused per-row top-3 over this wave's 64-col slab.
  // A given row's 64 values live entirely within one 16-lane half
  // (4 nf-fragments per lane), so a 4-step xor butterfly suffices.
  const float FINF = 3.0e38f;
  #pragma unroll
  for (int mf = 0; mf < 2; mf++) {
    #pragma unroll
    for (int r = 0; r < 8; r++) {
      const int row = rowBase + wm * 32 + mf * 16 + rsel + r;
      const float xn = xnorm[row];
      float d0 = FINF, d1 = FINF, d2 = FINF;
      int   i0 = -1,   i1 = -2,   i2 = -3;
      #pragma unroll
      for (int nf = 0; nf < 4; nf++) {
        const int c = colBase + wn * 64 + nf * 16 + lcol;
        const float dv = xn + enorm[c] - 2.0f * acc[mf][nf][r];
        top3_insert(dv, c, d0, i0, d1, i1, d2, i2);
      }
      #pragma unroll
      for (int m = 8; m; m >>= 1) {  // stays within the 16-lane half
        float e0 = __shfl_xor(d0, m, 32), e1 = __shfl_xor(d1, m, 32), e2 = __shfl_xor(d2, m, 32);
        int   j0 = __shfl_xor(i0, m, 32), j1 = __shfl_xor(i1, m, 32), j2 = __shfl_xor(i2, m, 32);
        top3_insert(e0, j0, d0, i0, d1, i1, d2, i2);
        top3_insert(e1, j1, d0, i0, d1, i1, d2, i2);
        top3_insert(e2, j2, d0, i0, d1, i1, d2, i2);
      }
      if (lcol < 3) {
        const float dsel = (lcol == 0) ? d0 : (lcol == 1) ? d1 : d2;
        const int   isel = (lcol == 0) ? i0 : (lcol == 1) ? i1 : i2;
        cand[(size_t)row * CSLOTS + blockIdx.x * 6 + wn * 3 + lcol] =
            make_float2(dsel, __int_as_float(isel));
      }
    }
  }
}

// ---------------------------------------------------------------------------
// Kernel 4: final top-3 from candidates + gather quant_st + EMA scatter.
// One wave per row; reads 192 candidates (1.5 KB) instead of 16 KB of dists.
// ---------------------------------------------------------------------------
__global__ __launch_bounds__(256) void k_topk(
    const float2* __restrict__ cand, const float* __restrict__ x,
    const float* __restrict__ emb,
    float* __restrict__ quant_st, float* __restrict__ idx_out,
    float* __restrict__ cnt, float* __restrict__ dw,
    float* __restrict__ loss_acc) {
  const int wave = (int)((blockIdx.x * blockDim.x + threadIdx.x) >> 5);
  const int lane = threadIdx.x & 31;
  if (wave >= Nv) return;

  const float FINF = 3.0e38f;
  float d0 = FINF, d1 = FINF, d2 = FINF;
  int   i0 = -1,   i1 = -2,   i2 = -3;

  const float2* crow = cand + (size_t)wave * CSLOTS;
  #pragma unroll
  for (int j = 0; j < CSLOTS / 32; j++) {
    const float2 c2 = crow[lane + j * 32];
    top3_insert(c2.x, __float_as_int(c2.y), d0, i0, d1, i1, d2, i2);
  }
  #pragma unroll
  for (int m = 16; m; m >>= 1) {
    float e0 = __shfl_xor(d0, m, 32), e1 = __shfl_xor(d1, m, 32), e2 = __shfl_xor(d2, m, 32);
    int   j0 = __shfl_xor(i0, m, 32), j1 = __shfl_xor(i1, m, 32), j2 = __shfl_xor(i2, m, 32);
    top3_insert(e0, j0, d0, i0, d1, i1, d2, i2);
    top3_insert(e1, j1, d0, i0, d1, i1, d2, i2);
    top3_insert(e2, j2, d0, i0, d1, i1, d2, i2);
  }
  // all 32 lanes now hold the identical global top-3 (ascending distance)

  if (lane < TOPK) {
    const int id = (lane == 0) ? i0 : (lane == 1) ? i1 : i2;
    idx_out[(size_t)wave * TOPK + lane] = (float)id;
    atomicAdd(&cnt[id], 1.0f);
  }

  const int ids[3] = { i0, i1, i2 };
  const float* xr = x + (size_t)wave * Dv;
  float lsum = 0.0f;
  #pragma unroll
  for (int j = 0; j < TOPK; j++) {
    const float* er = emb + (size_t)ids[j] * Dv;
    float* qo = quant_st + ((size_t)wave * TOPK + j) * Dv;
    float* dwr = dw + (size_t)ids[j] * Dv;
    #pragma unroll
    for (int v = 0; v < 2; v++) {
      const int c = v * 128 + lane * 4;
      const float4 ev = *(const float4*)(const void*)(er + c);
      const float4 xv = *(const float4*)(const void*)(xr + c);
      *(float4*)(void*)(qo + c) = ev;  // quant_st == quant numerically
      const float fx = ev.x - xv.x, fy = ev.y - xv.y, fz = ev.z - xv.z, fw = ev.w - xv.w;
      lsum += fx * fx + fy * fy + fz * fz + fw * fw;
      atomicAdd(dwr + c + 0, xv.x);   // dw = enc^T @ flat (scatter)
      atomicAdd(dwr + c + 1, xv.y);
      atomicAdd(dwr + c + 2, xv.z);
      atomicAdd(dwr + c + 3, xv.w);
    }
  }
  #pragma unroll
  for (int m = 16; m; m >>= 1) lsum += __shfl_xor(lsum, m, 32);
  if (lane == 0) atomicAdd(loss_acc, lsum);
}

// ---------------------------------------------------------------------------
// Kernel 5: finalize scalars + cluster-size normalization (single block)
// ---------------------------------------------------------------------------
__global__ __launch_bounds__(256) void k_finalize(
    const float* __restrict__ ema_cs, const float* __restrict__ cnt,
    const float* __restrict__ loss_acc,
    float* __restrict__ loss_out, float* __restrict__ perp_out,
    float* __restrict__ avg_probs_out, float* __restrict__ new_cs_out,
    float* __restrict__ cs_norm_ws) {
  __shared__ float s_n[256];
  __shared__ float s_h[256];
  float n_part = 0.0f, h_part = 0.0f;
  for (int k = (int)threadIdx.x; k < Kv; k += 256) {
    const float cs = ema_cs[k] * DECAYF + (1.0f - DECAYF) * cnt[k];
    n_part += cs;
    const float p = cnt[k] / (float)Nv;   // avg_probs = enc.mean(axis=0)
    avg_probs_out[k] = p;
    h_part += p * logf(p + 1e-10f);
  }
  s_n[threadIdx.x] = n_part;
  s_h[threadIdx.x] = h_part;
  __syncthreads();
  for (int s = 128; s; s >>= 1) {
    if ((int)threadIdx.x < s) {
      s_n[threadIdx.x] += s_n[threadIdx.x + s];
      s_h[threadIdx.x] += s_h[threadIdx.x + s];
    }
    __syncthreads();
  }
  const float n = s_n[0];
  for (int k = (int)threadIdx.x; k < Kv; k += 256) {
    const float cs  = ema_cs[k] * DECAYF + (1.0f - DECAYF) * cnt[k];
    const float csn = (cs + EPSF) / (n + (float)Kv * EPSF) * n;
    new_cs_out[k] = csn;
    cs_norm_ws[k] = csn;
  }
  if (threadIdx.x == 0) {
    *perp_out = expf(-s_h[0]);
    *loss_out = CCOST * (*loss_acc) / (float)((size_t)Nv * TOPK * Dv);
  }
}

// ---------------------------------------------------------------------------
// Kernel 6: EMA codebook update (K*D streaming, float4 vectorized)
// ---------------------------------------------------------------------------
__global__ __launch_bounds__(256) void k_emaw(
    const float* __restrict__ ema_w, const float* __restrict__ dw,
    const float* __restrict__ cs_norm,
    float* __restrict__ new_ema_w_out, float* __restrict__ upd_out) {
  const size_t i4 = (size_t)blockIdx.x * blockDim.x + threadIdx.x;
  if (i4 >= (size_t)Kv * Dv / 4) return;
  const float4 w4 = ((const float4*)ema_w)[i4];
  const float4 d4 = ((const float4*)dw)[i4];
  const float inv = 1.0f / cs_norm[(i4 * 4) / Dv];
  float4 nw;
  nw.x = w4.x * DECAYF + (1.0f - DECAYF) * d4.x;
  nw.y = w4.y * DECAYF + (1.0f - DECAYF) * d4.y;
  nw.z = w4.z * DECAYF + (1.0f - DECAYF) * d4.z;
  nw.w = w4.w * DECAYF + (1.0f - DECAYF) * d4.w;
  ((float4*)new_ema_w_out)[i4] = nw;
  float4 up;
  up.x = nw.x * inv; up.y = nw.y * inv; up.z = nw.z * inv; up.w = nw.w * inv;
  ((float4*)upd_out)[i4] = up;
}

// ---------------------------------------------------------------------------
extern "C" void kernel_launch(void* const* d_in, const int* in_sizes, int n_in,
                              void* d_out, int out_size, void* d_ws, size_t ws_size,
                              hipStream_t stream) {
  const float* x      = (const float*)d_in[0];   // [16,1024,256]
  const float* emb    = (const float*)d_in[1];   // [4096,256]
  const float* ema_cs = (const float*)d_in[2];   // [4096]
  const float* ema_w  = (const float*)d_in[3];   // [4096,256]

  // ---- output layout (reference tuple order, flat f32) ----
  float* out     = (float*)d_out;
  float* o_loss  = out;
  float* o_quant = out + 1;
  float* o_perp  = o_quant + (size_t)Nv * TOPK * Dv;
  float* o_avgp  = o_perp + 1;
  float* o_idx   = o_avgp + Kv;
  float* o_dist  = o_idx + (size_t)Nv * TOPK;
  float* o_ncs   = o_dist + (size_t)Nv * Kv;
  float* o_nemaw = o_ncs + Kv;
  float* o_upd   = o_nemaw + (size_t)Kv * Dv;

  // ---- workspace layout ----
  char* ws = (char*)d_ws;
  _Float16* xh = (_Float16*)ws;           ws += (size_t)Nv * Dv * sizeof(_Float16);
  _Float16* eh = (_Float16*)ws;           ws += (size_t)Kv * Dv * sizeof(_Float16);
  float* xnorm = (float*)ws;              ws += (size_t)Nv * sizeof(float);
  float* enorm = (float*)ws;              ws += (size_t)Kv * sizeof(float);
  float* zbase = (float*)ws;              // count | dw | loss (zeroed each call)
  float* cnt      = zbase;
  float* dw       = zbase + Kv;
  float* loss_acc = dw + (size_t)Kv * Dv;
  ws += ((size_t)Kv + (size_t)Kv * Dv + 1) * sizeof(float);
  float* cs_norm = (float*)ws;            ws += (size_t)Kv * sizeof(float);
  float2* cand   = (float2*)ws;           // Nv * 192 float2  (25 MB), fully
                                          // rewritten by k_dists every call

  // 1) zero the accumulators
  {
    const size_t nz = (size_t)Kv + (size_t)Kv * Dv + 1;
    k_zero<<<dim3((unsigned)((nz + 255) / 256)), dim3(256), 0, stream>>>(zbase, nz);
  }
  // 2) fp16 conversion + norms: one wave per row, (N+K) rows
  {
    const int waves = Nv + Kv;
    k_convert<<<dim3(waves / 8), dim3(256), 0, stream>>>(x, emb, xh, eh, xnorm, enorm);
  }
  // 3) WMMA distance GEMM + fused per-tile top-3 candidates
  k_dists<<<dim3(Kv / 128, Nv / 128), dim3(256), 0, stream>>>(xh, eh, xnorm, enorm,
                                                              o_dist, cand);
  // 4) final top-3 + gather + scatter stats: one wave per row
  k_topk<<<dim3(Nv / 8), dim3(256), 0, stream>>>(cand, x, emb, o_quant, o_idx,
                                                 cnt, dw, loss_acc);
  // 5) finalize scalars + normalized cluster sizes
  k_finalize<<<dim3(1), dim3(256), 0, stream>>>(ema_cs, cnt, loss_acc,
                                                o_loss, o_perp, o_avgp, o_ncs, cs_norm);
  // 6) EMA codebook update
  k_emaw<<<dim3((unsigned)(((size_t)Kv * Dv / 4 + 255) / 256)), dim3(256), 0, stream>>>(
      ema_w, dw, cs_norm, o_nemaw, o_upd);
}
